// MultiHeadAttention_63505386439213
// MI455X (gfx1250) — matrix operand order repaired and checked by reading, not compiled, not run on hardware
//
#include <hip/hip_runtime.h>

// MI455X / gfx1250, wave32, WMMA f32_16x16x32_f16 + TDM K-tile staging
typedef __attribute__((ext_vector_type(16))) _Float16 v16h;
typedef __attribute__((ext_vector_type(8)))  _Float16 h8;
typedef __attribute__((ext_vector_type(8)))  float    v8f;
typedef __attribute__((ext_vector_type(4)))  unsigned int u32x4;
typedef __attribute__((ext_vector_type(8)))  int      i32x8;
typedef __attribute__((ext_vector_type(4)))  int      i32x4;

#define B_   4
#define S_   2048
#define D_   1024
#define H_   16
#define DH_  64
#define MTOT (B_ * S_)   // 8192

#if defined(__has_builtin)
#if __has_builtin(__builtin_amdgcn_tensor_load_to_lds) && \
    __has_builtin(__builtin_amdgcn_s_wait_tensorcnt)
#define USE_TDM 1
#endif
#endif
#ifndef USE_TDM
#define USE_TDM 0
#endif

// ---------------------------------------------------------------------------
// Fragment helpers (layouts per CDNA5 ISA 7.12.2, wave32)
// ---------------------------------------------------------------------------
__device__ __forceinline__ h8 ld8(const _Float16* p) { return *(const h8*)p; }

// A-matrix 16x32 f16 fragment. Element (m,k) read from base[m*stride + kb + k].
__device__ __forceinline__ v16h make_afrag(const _Float16* base, int stride, int kb)
{
    const int lane = threadIdx.x & 31;
    const int hi   = lane >> 4;
    const int row  = lane & 15;
    const _Float16* rp = base + (size_t)row * stride + kb + 8 * hi;
    h8 c0 = ld8(rp);        // k = 8*hi .. 8*hi+7
    h8 c1 = ld8(rp + 16);   // k = 16+8*hi .. 16+8*hi+7
    v16h r;
#pragma unroll
    for (int i = 0; i < 8; ++i) { r[i] = c0[i]; r[i + 8] = c1[i]; }
    return r;
}

// B-matrix 32x16 f16 fragment where element (k,n) lives at base[n*stride + kb + k].
__device__ __forceinline__ v16h make_bfrag(const _Float16* base, int stride, int nb, int kb)
{
    const int lane = threadIdx.x & 31;
    const int hi   = lane >> 4;
    const int n    = nb + (lane & 15);
    const _Float16* rp = base + (size_t)n * stride + kb + 8 * hi;
    h8 c0 = ld8(rp);
    h8 c1 = ld8(rp + 16);
    v16h r;
#pragma unroll
    for (int i = 0; i < 8; ++i) { r[i] = c0[i]; r[i + 8] = c1[i]; }
    return r;
}

__device__ __forceinline__ v8f wmma_f16(v16h a, v16h b, v8f c)
{
    return __builtin_amdgcn_wmma_f32_16x16x32_f16(false, a, false, b,
                                                  (short)0, c, false, false);
}

__device__ __forceinline__ float hmax16(float v)
{
#pragma unroll
    for (int m = 1; m < 16; m <<= 1) v = fmaxf(v, __shfl_xor(v, m, 32));
    return v;
}
__device__ __forceinline__ float hsum16(float v)
{
#pragma unroll
    for (int m = 1; m < 16; m <<= 1) v += __shfl_xor(v, m, 32);
    return v;
}

// ---------------------------------------------------------------------------
// TDM: async copy of a [64 x 64-half] f16 tile, global -> LDS, with
// hardware row padding 64 -> 72 halves (pad_interval = 32 DW, pad_amount = 4 DW).
// All descriptor fields are wave-uniform.
// ---------------------------------------------------------------------------
#if USE_TDM
__device__ __forceinline__ void tdm_load_k_tile(const _Float16* gsrc,
                                                unsigned lds_byte_off,
                                                unsigned rows_remaining)
{
    const unsigned long long ga = (unsigned long long)(uintptr_t)gsrc;

    u32x4 g0;
    g0[0] = 1u;                                        // count = 1 valid D#
    g0[1] = lds_byte_off;                              // lds_addr
    g0[2] = (unsigned)(ga & 0xffffffffu);              // global_addr[31:0]
    g0[3] = (unsigned)((ga >> 32) & 0x01ffffffu)       // global_addr[56:32]
          | (2u << 30);                                // type = 2 ("image")

    i32x8 g1;
    g1[0] = (int)((1u << 16)      // data_size = 1 -> 2 bytes
                | (1u << 20)      // pad_enable
                | (4u << 22)      // pad_interval: 32 DWORDs (one 64-half row)
                | (3u << 25));    // pad_amount: 4 DWORDs (8 halves) -> stride 72
    g1[1] = (int)(64u << 16);                          // tensor_dim0 = 64 (dh)
    g1[2] = (int)((rows_remaining & 0xffffu) << 16);   // tensor_dim1 lo
    g1[3] = (int)(((rows_remaining >> 16) & 0xffffu)   // tensor_dim1 hi
                | (64u << 16));                        // tile_dim0 = 64
    g1[4] = 64;                                        // tile_dim1 = 64 keys
    g1[5] = 64;                                        // tensor_dim0_stride = 64
    g1[6] = 0;
    g1[7] = 0;

    i32x4 z4 = {};
#if defined(__clang_major__) && (__clang_major__ >= 23)
    i32x8 z8 = {};
    __builtin_amdgcn_tensor_load_to_lds(g0, g1, z4, z4, z8, 0);
#else
    __builtin_amdgcn_tensor_load_to_lds(g0, g1, z4, z4, 0);
#endif
}
#endif

// ---------------------------------------------------------------------------
// Flash attention. grid = (S/64, B*H); 128 threads = 4 waves; 64-key blocks.
// K tile staged by the Tensor Data Mover (wave 0) while all threads stage V^T.
// ---------------------------------------------------------------------------
#define KB_  64  // keys per block
#define SKT  72  // Kt row stride (halves) == 64 + TDM pad (8 halves)
#define SVT  72  // Vt row stride (halves), rows are dh,      cols are keys 0..63
#define SP   72  // P  row stride (halves), rows are queries, cols are keys 0..63

#define KT_HALVES (KB_ * SKT)
#define VT_HALVES (DH_ * SVT)
#define PB_HALVES (4 * 16 * SP)

__global__ __launch_bounds__(128)
void attn(const _Float16* __restrict__ qws, const _Float16* __restrict__ kws,
          const _Float16* __restrict__ vws, const float* __restrict__ scale_p,
          float* __restrict__ out)
{
    __shared__ __attribute__((aligned(128)))
        _Float16 lds_all[KT_HALVES + VT_HALVES + PB_HALVES];
    _Float16* Kt = lds_all;
    _Float16* Vt = lds_all + KT_HALVES;
    _Float16* Pb = lds_all + KT_HALVES + VT_HALVES;

    const int tid  = threadIdx.x;
    const int lane = tid & 31;
    const int w    = tid >> 5;
    const int hi   = lane >> 4;
    const int ln   = lane & 15;

    const int bh = blockIdx.y;
    const int b  = bh >> 4;
    const int h  = bh & 15;
    const size_t hb = (size_t)bh * S_ * DH_;
    const _Float16* Qh = qws + hb;
    const _Float16* Kh = kws + hb;
    const _Float16* Vh = vws + hb;
    const float rscale = 1.0f / scale_p[0];

    const int q0 = blockIdx.x * 64 + w * 16;

#if USE_TDM
    // LDS aperture: flat address low 32 bits == LDS byte offset (ISA 10.2)
    const unsigned kt_lds_off = (unsigned)(uintptr_t)(void*)Kt;
#endif

    // Q tile 16x64 as two A fragments, kept in registers
    const v16h qa0 = make_afrag(Qh + (size_t)q0 * DH_, DH_, 0);
    const v16h qa1 = make_afrag(Qh + (size_t)q0 * DH_, DH_, 32);

    v8f o0 = {}, o1 = {}, o2 = {}, o3 = {};
    float mi[8], li[8];
#pragma unroll
    for (int r = 0; r < 8; ++r) { mi[r] = -__builtin_inff(); li[r] = 0.0f; }

    _Float16* Pw = Pb + w * 16 * SP;   // wave-private P buffer

    for (int k0 = 0; k0 < S_; k0 += KB_) {
#if USE_TDM
        // wave 0: async TDM copy of K block [64 x 64] f16 -> Kt (stride 72)
        if (w == 0)
            tdm_load_k_tile(Kh + (size_t)k0 * DH_, kt_lds_off,
                            (unsigned)(S_ - k0));
#endif
        // all threads: stage V block transposed, Vt[dh][key]
#pragma unroll
        for (int j = 0; j < 4; ++j) {
            const int idx8 = tid + 128 * j;        // 512 chunks of 8 halves
            const int row  = idx8 >> 3;            // key 0..63
            const int col  = (idx8 & 7) * 8;       // dh  0,8,..,56
            const _Float16* vp = Vh + (size_t)(k0 + row) * DH_ + col;
            __builtin_prefetch(vp + KB_ * DH_, 0, 3);  // next key block
            h8 vv = ld8(vp);
#pragma unroll
            for (int e = 0; e < 8; ++e) Vt[(col + e) * SVT + row] = vv[e];
#if !USE_TDM
            h8 kv = ld8(Kh + (size_t)(k0 + row) * DH_ + col);
            *(h8*)&Kt[row * SKT + col] = kv;
#endif
        }
#if USE_TDM
        if (w == 0) __builtin_amdgcn_s_wait_tensorcnt(0);
#endif
        __syncthreads();

        // scores: 16 queries x 64 keys  (4 N-tiles x 2 K-steps)
        v8f s0 = {}, s1 = {}, s2 = {}, s3 = {};
        s0 = wmma_f16(qa0, make_bfrag(Kt, SKT,  0, 0), s0);
        s0 = wmma_f16(qa1, make_bfrag(Kt, SKT,  0, 32), s0);
        s1 = wmma_f16(qa0, make_bfrag(Kt, SKT, 16, 0), s1);
        s1 = wmma_f16(qa1, make_bfrag(Kt, SKT, 16, 32), s1);
        s2 = wmma_f16(qa0, make_bfrag(Kt, SKT, 32, 0), s2);
        s2 = wmma_f16(qa1, make_bfrag(Kt, SKT, 32, 32), s2);
        s3 = wmma_f16(qa0, make_bfrag(Kt, SKT, 48, 0), s3);
        s3 = wmma_f16(qa1, make_bfrag(Kt, SKT, 48, 32), s3);

        // online softmax over 64 columns (one reduction pass per 64 keys)
#pragma unroll
        for (int r = 0; r < 8; ++r) {
            const float a0 = s0[r] * rscale;
            const float a1 = s1[r] * rscale;
            const float a2 = s2[r] * rscale;
            const float a3 = s3[r] * rscale;
            const float rm   = hmax16(fmaxf(fmaxf(a0, a1), fmaxf(a2, a3)));
            const float mnew = fmaxf(mi[r], rm);
            const float al   = __expf(mi[r] - mnew);
            const float p0   = __expf(a0 - mnew);
            const float p1   = __expf(a1 - mnew);
            const float p2   = __expf(a2 - mnew);
            const float p3   = __expf(a3 - mnew);
            const float rs   = hsum16((p0 + p1) + (p2 + p3));
            li[r] = li[r] * al + rs;
            mi[r] = mnew;
            o0[r] *= al; o1[r] *= al; o2[r] *= al; o3[r] *= al;
            const int prow = (r + 8 * hi) * SP + ln;
            Pw[prow]      = (_Float16)p0;
            Pw[prow + 16] = (_Float16)p1;
            Pw[prow + 32] = (_Float16)p2;
            Pw[prow + 48] = (_Float16)p3;
        }

        // O += P(16x64) * V(64x64)   (wave-private P: no barrier needed)
        const v16h pa0 = make_afrag(Pw, SP, 0);
        const v16h pa1 = make_afrag(Pw, SP, 32);
        o0 = wmma_f16(pa0, make_bfrag(Vt, SVT,  0, 0), o0);
        o0 = wmma_f16(pa1, make_bfrag(Vt, SVT,  0, 32), o0);
        o1 = wmma_f16(pa0, make_bfrag(Vt, SVT, 16, 0), o1);
        o1 = wmma_f16(pa1, make_bfrag(Vt, SVT, 16, 32), o1);
        o2 = wmma_f16(pa0, make_bfrag(Vt, SVT, 32, 0), o2);
        o2 = wmma_f16(pa1, make_bfrag(Vt, SVT, 32, 32), o2);
        o3 = wmma_f16(pa0, make_bfrag(Vt, SVT, 48, 0), o3);
        o3 = wmma_f16(pa1, make_bfrag(Vt, SVT, 48, 32), o3);

        __syncthreads();   // protect Kt/Vt before next block overwrites
    }

    // normalize + store: out[b, s, h*64 + n]
#pragma unroll
    for (int r = 0; r < 8; ++r) {
        const float inv = 1.0f / li[r];
        const int srow  = q0 + r + 8 * hi;
        const size_t base = ((size_t)b * S_ + srow) * D_ + h * DH_ + ln;
        out[base +  0] = o0[r] * inv;
        out[base + 16] = o1[r] * inv;
        out[base + 32] = o2[r] * inv;
        out[base + 48] = o3[r] * inv;
    }
}

// ---------------------------------------------------------------------------
// QKV projection: Y[b,h,s,dh] (f16) = split_heads(X @ W^T + bias)
// 64x64 tile / 128 threads (4 waves), BK = 64.
// ---------------------------------------------------------------------------
#define SAP 72   // padded LDS row stride (halves)

__global__ __launch_bounds__(128)
void qkv_proj(const float* __restrict__ X, const float* __restrict__ W,
              const float* __restrict__ bias, _Float16* __restrict__ Y)
{
    __shared__ __attribute__((aligned(16))) _Float16 At[64 * SAP];
    __shared__ __attribute__((aligned(16))) _Float16 Wt[64 * SAP];

    const int tid  = threadIdx.x;
    const int lane = tid & 31;
    const int w    = tid >> 5;
    const int hi   = lane >> 4;
    const int mb   = blockIdx.x * 64;
    const int nb0  = blockIdx.y * 64;

    v8f acc[4] = {};

    for (int kb = 0; kb < D_; kb += 64) {
        // stage 64x64 fp32 tiles of X and W as f16 into LDS
#pragma unroll
        for (int j = 0; j < 8; ++j) {
            const int idx = tid + 128 * j;          // float4 index, 1024 total
            const int row = (idx * 4) >> 6;
            const int col = (idx * 4) & 63;
            const float* xp = X + (size_t)(mb + row) * D_ + kb + col;
            const float* wp = W + (size_t)(nb0 + row) * D_ + kb + col;
            // unconditional speculative prefetch of the next K-step tiles
            __builtin_prefetch(xp + 64, 0, 3);
            __builtin_prefetch(wp + 64, 0, 3);
            float4 xa = *(const float4*)xp;
            At[row * SAP + col + 0] = (_Float16)xa.x;
            At[row * SAP + col + 1] = (_Float16)xa.y;
            At[row * SAP + col + 2] = (_Float16)xa.z;
            At[row * SAP + col + 3] = (_Float16)xa.w;
            float4 wa = *(const float4*)wp;
            Wt[row * SAP + col + 0] = (_Float16)wa.x;
            Wt[row * SAP + col + 1] = (_Float16)wa.y;
            Wt[row * SAP + col + 2] = (_Float16)wa.z;
            Wt[row * SAP + col + 3] = (_Float16)wa.w;
        }
        __syncthreads();

        const v16h a0 = make_afrag(At + w * 16 * SAP, SAP, 0);
        const v16h a1 = make_afrag(At + w * 16 * SAP, SAP, 32);
#pragma unroll
        for (int t = 0; t < 4; ++t) {
            acc[t] = wmma_f16(a0, make_bfrag(Wt, SAP, t * 16, 0),  acc[t]);
            acc[t] = wmma_f16(a1, make_bfrag(Wt, SAP, t * 16, 32), acc[t]);
        }
        __syncthreads();
    }

    // epilogue: + bias, split heads, store f16
#pragma unroll
    for (int t = 0; t < 4; ++t) {
        const int n  = nb0 + t * 16 + (lane & 15);
        const float bv = bias[n];
        const int hh = n >> 6;
        const int dh = n & 63;
#pragma unroll
        for (int r = 0; r < 8; ++r) {
            const int m = mb + w * 16 + r + 8 * hi;
            const int b = m >> 11;
            const int s = m & (S_ - 1);
            const float v = acc[t][r] + bv;
            Y[((size_t)(b * H_ + hh) * S_ + s) * DH_ + dh] = (_Float16)v;
        }
    }
}

// ---------------------------------------------------------------------------
extern "C" void kernel_launch(void* const* d_in, const int* in_sizes, int n_in,
                              void* d_out, int out_size, void* d_ws, size_t ws_size,
                              hipStream_t stream)
{
    (void)in_sizes; (void)n_in; (void)out_size; (void)ws_size;

    const float* query = (const float*)d_in[0];
    const float* key   = (const float*)d_in[1];
    const float* value = (const float*)d_in[2];
    const float* Wq    = (const float*)d_in[3];
    const float* bq    = (const float*)d_in[4];
    const float* Wk    = (const float*)d_in[5];
    const float* bk    = (const float*)d_in[6];
    const float* Wv    = (const float*)d_in[7];
    const float* bv    = (const float*)d_in[8];
    const float* scale = (const float*)d_in[9];
    float* out = (float*)d_out;

    const size_t N = (size_t)B_ * H_ * S_ * DH_;
    _Float16* qws = (_Float16*)d_ws;
    _Float16* kws = qws + N;
    _Float16* vws = kws + N;

    dim3 gp(MTOT / 64, D_ / 64);
    qkv_proj<<<gp, 128, 0, stream>>>(query, Wq, bq, qws);
    qkv_proj<<<gp, 128, 0, stream>>>(key,   Wk, bk, kws);
    qkv_proj<<<gp, 128, 0, stream>>>(value, Wv, bv, vws);

    dim3 ga(S_ / 64, B_ * H_);
    attn<<<ga, 128, 0, stream>>>(qws, kws, vws, scale, out);
}
